// FasterRCNN_72078141161687
// MI455X (gfx1250) — compile-verified, matrix operand order checked by reading, not run on hardware
//
#include <hip/hip_runtime.h>
#include <hip/hip_bf16.h>
#include <math.h>

// ---------------------------------------------------------------------------
// Faster R-CNN ROI head for MI455X (gfx1250, wave32, WMMA + async-to-LDS).
//
// FC GEMM M=128 x K=50176 x N=1024 dominates. fc_w converted to bf16 (103MB,
// L2-resident), GEMM on V_WMMA_F32_16X16X32_BF16 with:
//   - 32x32 register tile per wave (4 accumulators, 4 wmma per k-step)
//   - A tile (shared by all 8 waves of a block) staged via
//     GLOBAL_LOAD_ASYNC_TO_LDS_B64 + S_WAIT_ASYNCCNT, double-buffered in LDS
//   - K split x32 for occupancy, partials reduced in a follow-up pass
// ---------------------------------------------------------------------------

#define C_CH   1024
#define H_DIM  100
#define W_DIM  50
#define P_DIM  7
#define N_ROI  128
#define K_FC   (C_CH * P_DIM * P_DIM)   // 50176
#define OUT_FC 1024
#define NCLS   21
#define KSPLIT 32
#define KCHUNK (K_FC / KSPLIT)          // 1568
#define KITER  (KCHUNK / 32)            // 49
#define LDS_ROW_STRIDE 80               // 64B of K data + 16B pad (bank spread)
#define LDS_TILE_BYTES (32 * LDS_ROW_STRIDE)  // 2560B per buffer

typedef __attribute__((ext_vector_type(16))) __bf16        v16bf;
typedef __attribute__((ext_vector_type(8)))  float         v8f;
typedef __attribute__((ext_vector_type(4)))  unsigned int  v4u;

union FragBF { v16bf v; v4u u[2]; };

__device__ __forceinline__ unsigned short f2bf(float f) {
    unsigned int u = __builtin_bit_cast(unsigned int, f);
    unsigned int r = u + 0x7FFFu + ((u >> 16) & 1u);   // RNE
    return (unsigned short)(r >> 16);
}

__device__ __forceinline__ void async_copy_b64(unsigned lds_off, const void* gptr) {
    unsigned long long ga = (unsigned long long)(size_t)gptr;
    asm volatile("global_load_async_to_lds_b64 %0, %1, off"
                 :: "v"(lds_off), "v"(ga) : "memory");
}

__device__ __forceinline__ void wait_async0() {
    asm volatile("s_wait_asynccnt 0x0" ::: "memory");
}

// ---- Kernel 0: fc_w fp32 -> bf16 (205MB read once; bf16 copy fits in 192MB L2)
__global__ __launch_bounds__(256) void convert_w_bf16(const float* __restrict__ w,
                                                      unsigned short* __restrict__ o) {
    long long i = (long long)blockIdx.x * 256 + threadIdx.x;
    long long base = i * 8;
    union { v4u u; unsigned short s[8]; } pack;
#pragma unroll
    for (int k = 0; k < 8; ++k) pack.s[k] = f2bf(w[base + k]);
    *(v4u*)(o + base) = pack.u;
}

// ---- Kernel 1: ROI crop + adaptive max pool -> bf16 activations [N, 50176]
__global__ __launch_bounds__(256) void roi_pool(const float* __restrict__ fmap,
                                                const int* __restrict__ rx,
                                                const int* __restrict__ ry,
                                                unsigned short* __restrict__ pooled) {
    const int n = blockIdx.x;
    const int cbase = blockIdx.y * 32;
    const int xlo = rx[2 * n], xhi = rx[2 * n + 1];   // rows (H axis)
    const int ylo = ry[2 * n], yhi = ry[2 * n + 1];   // cols (W axis; may exceed W, matching ref)
    const int nx = xhi - xlo + 1;
    const int ny = yhi - ylo + 1;
    for (int wi = threadIdx.x; wi < 32 * 49; wi += 256) {
        const int cl  = wi / 49;
        const int bin = wi % 49;
        const int i = bin / 7, j = bin % 7;
        int rs = xlo + (i * nx) / 7;
        int re = xlo + ((i + 1) * nx + 6) / 7;
        int cs = ylo + (j * ny) / 7;
        int ce = ylo + ((j + 1) * ny + 6) / 7;
        if (rs < 0) rs = 0;  if (re > H_DIM) re = H_DIM;
        if (cs < 0) cs = 0;  if (ce > W_DIM) ce = W_DIM;
        float m = -3.4028234663852886e38f;
        const int c = cbase + cl;
        const float* f = fmap + (size_t)c * H_DIM * W_DIM;
        for (int r = rs; r < re; ++r)
            for (int col = cs; col < ce; ++col) {
                float v = f[r * W_DIM + col];
                m = v > m ? v : m;
            }
        pooled[(size_t)n * K_FC + (size_t)c * 49 + bin] = f2bf(m);
    }
}

// ---- Kernel 2: bf16 WMMA GEMM, 32x32 tile per wave, async A-staging in LDS.
// grid (M/32=4, N/(8*32)=4, KSPLIT), block 256 = 8 waves (waves split N).
// A frag layout (ISA 7.12.2, 16-bit 16x32): lane<16 -> K{0..7,16..23},
// lane>=16 -> K{8..15,24..31} of row (lane&15); fp32 C/D: VGPR r -> M=r+8*half.
__global__ __launch_bounds__(256) void gemm_fc_wmma(const unsigned short* __restrict__ A,
                                                    const unsigned short* __restrict__ B,
                                                    float* __restrict__ partial) {
    __shared__ __attribute__((aligned(16))) unsigned char atile[2][LDS_TILE_BYTES];

    const int tid  = threadIdx.x;
    const int lane = tid & 31;
    const int wave = tid >> 5;
    const int l    = lane & 15;
    const int half = lane >> 4;
    const int mbase = blockIdx.x * 32;
    const int o0    = (blockIdx.y * 8 + wave) * 32;
    const int kb0   = blockIdx.z * KCHUNK;

    // staging role: thread -> (row r of 32, 8-byte segment seg of 8) covering 32x32 bf16
    const int r   = tid >> 3;
    const int seg = tid & 7;
    const unsigned char* agp =
        (const unsigned char*)(A + (size_t)(mbase + r) * K_FC + kb0) + seg * 8;
    const unsigned lds_base = (unsigned)(size_t)(&atile[0][0]);
    const unsigned lds_thr  = r * LDS_ROW_STRIDE + seg * 8;

    // B pointers: two 16-col fragments per wave
    const unsigned short* bp0 = B + (size_t)(o0 + l) * K_FC + kb0 + half * 8;
    const unsigned short* bp1 = B + (size_t)(o0 + 16 + l) * K_FC + kb0 + half * 8;

    v8f c00 = {0.f,0.f,0.f,0.f,0.f,0.f,0.f,0.f};
    v8f c01 = c00, c10 = c00, c11 = c00;

    // prologue: stage k-step 0 into buffer 0
    async_copy_b64(lds_base + lds_thr, agp);

    for (int it = 0; it < KITER; ++it) {
        const int buf = it & 1;
        wait_async0();
        __syncthreads();                       // atile[buf] ready; prior reads of buf^1 done
        if (it + 1 < KITER)
            async_copy_b64(lds_base + (buf ^ 1) * LDS_TILE_BYTES + lds_thr,
                           agp + (size_t)(it + 1) * 64);

        const unsigned char* at = &atile[buf][0];
        FragBF a0, a1, b0, b1;
        a0.u[0] = *(const v4u*)(at + l * LDS_ROW_STRIDE + half * 16);
        a0.u[1] = *(const v4u*)(at + l * LDS_ROW_STRIDE + 32 + half * 16);
        a1.u[0] = *(const v4u*)(at + (16 + l) * LDS_ROW_STRIDE + half * 16);
        a1.u[1] = *(const v4u*)(at + (16 + l) * LDS_ROW_STRIDE + 32 + half * 16);
        b0.u[0] = *(const v4u*)(bp0);
        b0.u[1] = *(const v4u*)(bp0 + 16);
        b1.u[0] = *(const v4u*)(bp1);
        b1.u[1] = *(const v4u*)(bp1 + 16);
        bp0 += 32; bp1 += 32;

        c00 = __builtin_amdgcn_wmma_f32_16x16x32_bf16(false, a0.v, false, b0.v, (short)0, c00, false, false);
        c01 = __builtin_amdgcn_wmma_f32_16x16x32_bf16(false, a0.v, false, b1.v, (short)0, c01, false, false);
        c10 = __builtin_amdgcn_wmma_f32_16x16x32_bf16(false, a1.v, false, b0.v, (short)0, c10, false, false);
        c11 = __builtin_amdgcn_wmma_f32_16x16x32_bf16(false, a1.v, false, b1.v, (short)0, c11, false, false);
    }

    float* out = partial + (size_t)blockIdx.z * N_ROI * OUT_FC;
    const int gn0 = o0 + l, gn1 = o0 + 16 + l;
    const int mr0 = mbase + half * 8, mr1 = mbase + 16 + half * 8;
#pragma unroll
    for (int rr = 0; rr < 8; ++rr) {
        out[(size_t)(mr0 + rr) * OUT_FC + gn0] = c00[rr];
        out[(size_t)(mr0 + rr) * OUT_FC + gn1] = c01[rr];
        out[(size_t)(mr1 + rr) * OUT_FC + gn0] = c10[rr];
        out[(size_t)(mr1 + rr) * OUT_FC + gn1] = c11[rr];
    }
}

// ---- Kernel 3: sum K-split partials + bias -> r [128,1024] fp32
__global__ __launch_bounds__(256) void reduce_bias(const float* __restrict__ partial,
                                                   const float* __restrict__ bias,
                                                   float* __restrict__ r) {
    const int idx = blockIdx.x * 256 + threadIdx.x;    // 0..131071
    const int o = idx & (OUT_FC - 1);
    float s = bias[o];
#pragma unroll
    for (int k = 0; k < KSPLIT; ++k)
        s += partial[(size_t)k * N_ROI * OUT_FC + idx];
    r[idx] = s;
}

// ---- Kernel 4: cls/reg heads + softmax. One block per ROI, wave32 shuffles.
__global__ __launch_bounds__(256) void heads(const float* __restrict__ r,
                                             const float* __restrict__ cls_w,
                                             const float* __restrict__ cls_b,
                                             const float* __restrict__ reg_w,
                                             const float* __restrict__ reg_b,
                                             float* __restrict__ out) {
    __shared__ float row[OUT_FC];
    __shared__ float logits[32];
    const int n = blockIdx.x;
    for (int k = threadIdx.x; k < OUT_FC; k += 256) row[k] = r[(size_t)n * OUT_FC + k];
    __syncthreads();
    const int wave = threadIdx.x >> 5, lane = threadIdx.x & 31;
    for (int o = wave; o < NCLS + 4; o += 8) {
        const float* w = (o < NCLS) ? (cls_w + (size_t)o * OUT_FC)
                                    : (reg_w + (size_t)(o - NCLS) * OUT_FC);
        float s = 0.f;
        for (int k = lane; k < OUT_FC; k += 32) s += row[k] * w[k];
#pragma unroll
        for (int off = 16; off > 0; off >>= 1) s += __shfl_xor(s, off, 32);
        if (lane == 0) logits[o] = s + ((o < NCLS) ? cls_b[o] : reg_b[o - NCLS]);
    }
    __syncthreads();
    if (threadIdx.x == 0) {
        float mx = logits[0];
        for (int j = 1; j < NCLS; ++j) mx = logits[j] > mx ? logits[j] : mx;
        float e[NCLS]; float sum = 0.f;
        for (int j = 0; j < NCLS; ++j) { e[j] = expf(logits[j] - mx); sum += e[j]; }
        const float inv = 1.f / sum;
        for (int j = 0; j < NCLS; ++j) out[(size_t)n * NCLS + j] = e[j] * inv;
        for (int j = 0; j < 4; ++j)    out[N_ROI * NCLS + (size_t)n * 4 + j] = logits[NCLS + j];
    }
}

extern "C" void kernel_launch(void* const* d_in, const int* in_sizes, int n_in,
                              void* d_out, int out_size, void* d_ws, size_t ws_size,
                              hipStream_t stream) {
    const float* feature_map = (const float*)d_in[0];
    const float* fc_w        = (const float*)d_in[1];
    const float* fc_b        = (const float*)d_in[2];
    const float* cls_w       = (const float*)d_in[3];
    const float* cls_b       = (const float*)d_in[4];
    const float* reg_w       = (const float*)d_in[5];
    const float* reg_b       = (const float*)d_in[6];
    const int*   rois_x      = (const int*)d_in[7];
    const int*   rois_y      = (const int*)d_in[8];
    float* out = (float*)d_out;

    // Workspace layout (all regions rewritten every launch -> deterministic)
    char* ws = (char*)d_ws;
    unsigned short* wbf    = (unsigned short*)ws;                                   // 102,760,448 B
    unsigned short* pooled = (unsigned short*)(ws + (size_t)102760448);             //  12,845,056 B
    float*          part   = (float*)(ws + (size_t)102760448 + 12845056);           //  16,777,216 B
    float*          rbuf   = (float*)(ws + (size_t)102760448 + 12845056 + 16777216); //    524,288 B

    convert_w_bf16<<<(OUT_FC * (long long)K_FC) / (8 * 256), 256, 0, stream>>>(fc_w, wbf);
    roi_pool<<<dim3(N_ROI, C_CH / 32), 256, 0, stream>>>(feature_map, rois_x, rois_y, pooled);
    gemm_fc_wmma<<<dim3(N_ROI / 32, OUT_FC / (8 * 32), KSPLIT), 256, 0, stream>>>(pooled, wbf, part);
    reduce_bias<<<(N_ROI * OUT_FC) / 256, 256, 0, stream>>>(part, fc_b, rbuf);
    heads<<<N_ROI, 256, 0, stream>>>(rbuf, cls_w, cls_b, reg_w, reg_b, out);
}
